// SupConLoss_86105504350689
// MI455X (gfx1250) — matrix operand order repaired and checked by reading, not compiled, not run on hardware
//
#include <hip/hip_runtime.h>
#include <hip/hip_bf16.h>

// ---------------------------------------------------------------------------
// SupCon loss, M=8192, K=128, T=0.2, all proposals kept, coef=1.
//   loss_i = -[ (S_i - n_i*mx_i) - n_i*log(max(Z_i,1e-5)) ] / (n_i + 1)
//   S_i  = sum_{j!=i, lab match} sim_ij          (raw sim, pre-shift)
//   mx_i = max_j sim_ij (incl diagonal)
//   Z_i  = [sum_{j!=i} exp(sim_ij - 5)] * exp(5 - mx_i)
//   out  = mean_i loss_i
// ---------------------------------------------------------------------------

#define MROWS   8192
#define KDIM    128
#define INV_T   5.0f
#define CSPLIT  8                         // column splits (grid.x)
#define CTILES  ((MROWS / CSPLIT) / 16)   // 64 column tiles of 16 per split
#define RBLK    128                       // rows per block (8 waves * 16)

typedef _Float16 h16 __attribute__((ext_vector_type(16)));
typedef _Float16 h8  __attribute__((ext_vector_type(8)));
typedef _Float16 h4  __attribute__((ext_vector_type(4)));
typedef float    f8  __attribute__((ext_vector_type(8)));
typedef int      v4i __attribute__((ext_vector_type(4)));

typedef __attribute__((address_space(1))) v4i gv4i;   // global int4
typedef __attribute__((address_space(3))) v4i lv4i;   // LDS int4

union H16U { h16 v; h8 h[2]; };

#if __has_builtin(__builtin_amdgcn_global_load_async_to_lds_b128) && \
    __has_builtin(__builtin_amdgcn_s_wait_asynccnt)
#define HAVE_ASYNC_LDS 1
#else
#define HAVE_ASYNC_LDS 0
#endif

// ---------------------------------------------------------------------------
// Kernel 1: row-wise L2 normalize, f32 -> f16.  One wave per row.
// ---------------------------------------------------------------------------
__global__ __launch_bounds__(256)
void supcon_prep(const float* __restrict__ feat, _Float16* __restrict__ fn) {
    const int row  = blockIdx.x * 8 + (threadIdx.x >> 5);
    const int lane = threadIdx.x & 31;
    const float4 v = ((const float4*)(feat + (size_t)row * KDIM))[lane];
    float ss = v.x * v.x + v.y * v.y + v.z * v.z + v.w * v.w;
#pragma unroll
    for (int m = 16; m >= 1; m >>= 1) ss += __shfl_xor(ss, m, 32);
    const float inv = 1.0f / sqrtf(ss);
    h4 o;
    o[0] = (_Float16)(v.x * inv);
    o[1] = (_Float16)(v.y * inv);
    o[2] = (_Float16)(v.z * inv);
    o[3] = (_Float16)(v.w * inv);
    ((h4*)(fn + (size_t)row * KDIM))[lane] = o;
}

// ---------------------------------------------------------------------------
// Stage one 16-col x 128-K f16 B tile (4 KB) into LDS: 256 threads, one
// 16-byte chunk each.  LDS layout = row-major [col][K], byte offset = t*16.
// ---------------------------------------------------------------------------
__device__ __forceinline__ void stage_tile(const _Float16* __restrict__ fn,
                                           int colbase, _Float16* dst, int t) {
    const _Float16* g = fn + (size_t)(colbase + (t >> 4)) * KDIM + (t & 15) * 8;
#if HAVE_ASYNC_LDS
    __builtin_amdgcn_global_load_async_to_lds_b128(
        (gv4i*)g, (lv4i*)(dst + (size_t)t * 8), 0, 0);
#else
    const float4 v = *(const float4*)g;
    *(float4*)(dst + (size_t)t * 8) = v;
#endif
}

// ---------------------------------------------------------------------------
// Kernel 2: fused  f f^T  WMMA + streaming row reductions.
// grid = (CSPLIT, MROWS/RBLK); block = 256 (8 waves, each owns 16 rows).
// B tiles staged once per block into LDS (double-buffered, async DMA),
// consumed by all 8 waves via ds_load_b128.
// ---------------------------------------------------------------------------
__global__ __launch_bounds__(256)
void supcon_main(const _Float16* __restrict__ fn, const int* __restrict__ labels,
                 float4* __restrict__ part) {
    __shared__ __align__(32) _Float16 bbuf[2][16 * KDIM];   // 2 x 4 KB

    const int t       = threadIdx.x;
    const int lane    = t & 31;
    const int wave    = t >> 5;
    const int laneHi  = lane >> 4;     // 0: lanes 0-15, 1: lanes 16-31
    const int lane16  = lane & 15;
    const int rowbase = blockIdx.y * RBLK + wave * 16;
    const int colbeg  = blockIdx.x * (MROWS / CSPLIT);

    // Resident A fragments: 16 rows x 128 K of f16 (32 VGPRs).
    const h8* arow = (const h8*)(fn + (size_t)(rowbase + lane16) * KDIM);
    H16U a[4];
#pragma unroll
    for (int kb = 0; kb < 4; ++kb) {
        a[kb].h[0] = arow[kb * 4 + 2 * laneHi];       // K = kb*32 + hi*16 .. +7
        a[kb].h[1] = arow[kb * 4 + 2 * laneHi + 1];   // K = kb*32 + hi*16 + 8 .. +15
    }

    // Labels of the 8 rows this lane's C elements belong to.
    int rlab[8];
#pragma unroll
    for (int v = 0; v < 8; ++v) rlab[v] = labels[rowbase + 8 * laneHi + v];

    float mx[8], se[8], ms[8], mc[8];
#pragma unroll
    for (int v = 0; v < 8; ++v) { mx[v] = -1e30f; se[v] = 0.f; ms[v] = 0.f; mc[v] = 0.f; }

    stage_tile(fn, colbeg, bbuf[0], t);           // prologue: tile 0 in flight

    for (int ct = 0; ct < CTILES; ++ct) {
        const int cur     = ct & 1;
        const int colbase = colbeg + ct * 16;
        const int mycol   = colbase + lane16;

        if (ct + 1 < CTILES) {
            stage_tile(fn, colbase + 16, bbuf[cur ^ 1], t);
            if (t < 16 && ct + 2 < CTILES)        // pull tile ct+2 toward L2/WGP
                __builtin_prefetch(fn + (size_t)(colbase + 32 + t) * KDIM, 0, 3);
#if HAVE_ASYNC_LDS
            __builtin_amdgcn_s_wait_asynccnt(1);  // current buffer landed,
#endif                                            // next tile still in flight
        } else {
#if HAVE_ASYNC_LDS
            __builtin_amdgcn_s_wait_asynccnt(0);  // drain on last iteration
#endif
        }
        __syncthreads();                          // bbuf[cur] ready for all waves

        // B fragments from LDS: lane holds 16 contiguous K of one column.
        const h16* bcol = (const h16*)(bbuf[cur] + lane16 * KDIM);
        const h16 b0 = bcol[0 + laneHi];
        const h16 b1 = bcol[2 + laneHi];
        const h16 b2 = bcol[4 + laneHi];
        const h16 b3 = bcol[6 + laneHi];
        const int clab = labels[mycol];

        f8 c = {0.f, 0.f, 0.f, 0.f, 0.f, 0.f, 0.f, 0.f};
        c = __builtin_amdgcn_wmma_f32_16x16x32_f16(false, a[0].v, false, b0, (short)0, c, false, false);
        c = __builtin_amdgcn_wmma_f32_16x16x32_f16(false, a[1].v, false, b1, (short)0, c, false, false);
        c = __builtin_amdgcn_wmma_f32_16x16x32_f16(false, a[2].v, false, b2, (short)0, c, false, false);
        c = __builtin_amdgcn_wmma_f32_16x16x32_f16(false, a[3].v, false, b3, (short)0, c, false, false);

#pragma unroll
        for (int v = 0; v < 8; ++v) {
            const int rowg = rowbase + 8 * laneHi + v;
            const float s  = c[v] * INV_T;                   // cosine / T
            mx[v] = fmaxf(mx[v], s);                         // max incl. diagonal
            const bool offd  = (mycol != rowg);
            const float e    = __expf(s - INV_T);            // fixed shift by 1/T
            se[v] += offd ? e : 0.0f;
            const bool match = offd && (clab == rlab[v]);
            ms[v] += match ? s : 0.0f;
            mc[v] += match ? 1.0f : 0.0f;
        }

        __syncthreads();   // all waves done with bbuf[cur] before it is re-staged
    }

    // Reduce each row across its 16 lanes (xor masks stay within each half).
#pragma unroll
    for (int m = 1; m <= 8; m <<= 1) {
#pragma unroll
        for (int v = 0; v < 8; ++v) {
            mx[v] = fmaxf(mx[v], __shfl_xor(mx[v], m, 32));
            se[v] += __shfl_xor(se[v], m, 32);
            ms[v] += __shfl_xor(ms[v], m, 32);
            mc[v] += __shfl_xor(mc[v], m, 32);
        }
    }

    if (lane16 == 0) {
#pragma unroll
        for (int v = 0; v < 8; ++v) {
            const int row = rowbase + 8 * laneHi + v;
            part[(size_t)row * CSPLIT + blockIdx.x] = make_float4(mx[v], se[v], ms[v], mc[v]);
        }
    }
}

// ---------------------------------------------------------------------------
// Kernel 3: combine the CSPLIT partials per row, finish the loss, write mean.
// ---------------------------------------------------------------------------
__global__ __launch_bounds__(256)
void supcon_final(const float4* __restrict__ part, float* __restrict__ out) {
    __shared__ float red[256];
    float acc = 0.0f;
    for (int r = threadIdx.x; r < MROWS; r += 256) {
        float MX = -1e30f, SE = 0.f, MS = 0.f, MC = 0.f;
#pragma unroll
        for (int s = 0; s < CSPLIT; ++s) {
            const float4 p = part[(size_t)r * CSPLIT + s];
            MX = fmaxf(MX, p.x); SE += p.y; MS += p.z; MC += p.w;
        }
        float Z = SE * __expf(INV_T - MX);     // re-shift: 5 -> true row max
        Z = fmaxf(Z, 1e-5f);                   // clip before log (as reference)
        const float logZ = logf(Z);
        const float num  = (MS - MC * MX) - MC * logZ;
        acc += -num / (MC + 1.0f);
    }
    red[threadIdx.x] = acc;
    __syncthreads();
    for (int s = 128; s > 0; s >>= 1) {
        if (threadIdx.x < s) red[threadIdx.x] += red[threadIdx.x + s];
        __syncthreads();
    }
    if (threadIdx.x == 0) out[0] = red[0] / (float)MROWS;
}

// ---------------------------------------------------------------------------
extern "C" void kernel_launch(void* const* d_in, const int* in_sizes, int n_in,
                              void* d_out, int out_size, void* d_ws, size_t ws_size,
                              hipStream_t stream) {
    (void)in_sizes; (void)n_in; (void)out_size; (void)ws_size;
    const float* feat   = (const float*)d_in[0];
    // d_in[1] = ious: coef = 1 for reweight 'none' -> unused.
    const int*   labels = (const int*)d_in[2];

    _Float16* fn   = (_Float16*)d_ws;                                   // 2 MB
    float4*   part = (float4*)((char*)d_ws + (size_t)MROWS * KDIM * 2); // 1 MB

    supcon_prep <<<MROWS / 8, 256, 0, stream>>>(feat, fn);
    supcon_main <<<dim3(CSPLIT, MROWS / RBLK), 256, 0, stream>>>(fn, labels, part);
    supcon_final<<<1, 256, 0, stream>>>(part, (float*)d_out);
}
